// QuantLinear_9019431321834
// MI455X (gfx1250) — matrix operand order repaired and checked by reading, not compiled, not run on hardware
//
#include <hip/hip_runtime.h>

typedef __attribute__((ext_vector_type(16))) _Float16 v16h;
typedef __attribute__((ext_vector_type(8)))  _Float16 v8h;
typedef __attribute__((ext_vector_type(8)))  float    v8f;

#define INF    4096
#define OUTF   11008
#define NROWS  8
#define NTILES 86      // 11008 / 128 columns per block
#define KSPLIT 4       // K split 4 ways -> 344 blocks
#define KSLICE 1024    // 4096 / 4
#define TITERS 32      // 1024 / 32 k per WMMA step
#define XWS_BYTES (128 * 512)   // full-K fragment-ordered x as f16: 64 KB

// out[r, o] = bias[o]  (split-K partials are atomically accumulated on top)
__global__ __launch_bounds__(256) void bias_init_kernel(const float* __restrict__ bias,
                                                        float* __restrict__ out) {
    int i = blockIdx.x * 256 + threadIdx.x;
    if (i < NROWS * OUTF) {
        int o = i - (i / OUTF) * OUTF;
        out[i] = bias[o];
    }
}

// One-shot: convert x[8,4096] f32 -> f16 in WMMA-A fragment order.
// Global layout: chunk c (0..4095) of 16B at offset c*16, where
// c = t*32 + part*16 + slot, slot = g*8 + m, covering x[m][t*32+part*16+g*8 .. +7].
__global__ __launch_bounds__(256) void xconvert_kernel(const float* __restrict__ x,
                                                       _Float16* __restrict__ xf16) {
    int c    = blockIdx.x * 256 + threadIdx.x;     // 4096 chunks total
    int t    = c >> 5;
    int rem  = c & 31;
    int part = rem >> 4;
    int slot = rem & 15;
    int g    = slot >> 3;
    int m    = slot & 7;
    int k0   = t * 32 + part * 16 + g * 8;
    const float4* px = (const float4*)(x + m * INF + k0);
    float4 f0 = px[0], f1 = px[1];
    v8h h;
    h[0] = (_Float16)f0.x; h[1] = (_Float16)f0.y;
    h[2] = (_Float16)f0.z; h[3] = (_Float16)f0.w;
    h[4] = (_Float16)f1.x; h[5] = (_Float16)f1.y;
    h[6] = (_Float16)f1.z; h[7] = (_Float16)f1.w;
    *(v8h*)((char*)xf16 + (size_t)c * 16) = h;
}

template <bool ASYNC_STAGE>
__global__ __launch_bounds__(256) void q4gemm_wmma_kernel(
    const float*    __restrict__ x,        // [8, 4096] f32 (fallback path)
    const _Float16* __restrict__ xf16,     // fragment-ordered f16 in d_ws (fast path)
    const unsigned* __restrict__ qweight,  // [512, 11008] packed int4
    const float*    __restrict__ scales,   // [11008]
    const int*      __restrict__ zeros,    // [5504], two 4-bit zps per entry
    float*          __restrict__ out)      // [8, 11008]
{
    extern __shared__ char smem[];         // 16 KB x-slice, fragment-ordered
    const int tid    = threadIdx.x;
    const int ntile  = blockIdx.x % NTILES;
    const int kslice = blockIdx.x / NTILES;
    const int kbase  = kslice * KSLICE;

    if (ASYNC_STAGE) {
        // Pure DMA: 16 KB of pre-swizzled f16 x straight into LDS, ASYNCcnt-tracked.
        const char* src = (const char*)xf16 + (size_t)kslice * (TITERS * 512);
        #pragma unroll
        for (int c = tid; c < TITERS * 32; c += 256) {
            int lo = c * 16;
            asm volatile("global_load_async_to_lds_b128 %0, %1, off"
                         :: "v"(lo), "v"(src + (size_t)lo) : "memory");
        }
        asm volatile("s_wait_asynccnt 0" ::: "memory");
    } else {
        // Fallback: convert f32 x slice in-kernel into the same LDS layout.
        for (int c = tid; c < TITERS * 32; c += 256) {
            int t    = c >> 5;
            int rem  = c & 31;
            int part = rem >> 4;
            int slot = rem & 15;
            int g    = slot >> 3;
            int m    = slot & 7;
            int k0   = kbase + t * 32 + part * 16 + g * 8;
            const float4* px = (const float4*)(x + m * INF + k0);
            float4 f0 = px[0], f1 = px[1];
            v8h h;
            h[0] = (_Float16)f0.x; h[1] = (_Float16)f0.y;
            h[2] = (_Float16)f0.z; h[3] = (_Float16)f0.w;
            h[4] = (_Float16)f1.x; h[5] = (_Float16)f1.y;
            h[6] = (_Float16)f1.z; h[7] = (_Float16)f1.w;
            *(v8h*)(smem + t * 512 + part * 256 + slot * 16) = h;
        }
    }
    __syncthreads();

    const int wave = tid >> 5;
    const int lane = tid & 31;
    const int g    = lane >> 4;          // lane group (k-half selector)
    const int m8   = lane & 7;           // A rows 8-15 are don't-care: alias rows 0-7
    const int n    = ntile * 128 + wave * 16 + (lane & 15);

    // Per-column dequant constants: w = q*s - z*s
    const float    sf  = scales[n];
    const unsigned zc  = (unsigned)zeros[n >> 1];
    const float    zf  = (float)((zc >> (4 * (n & 1))) & 0xFu);
    const float    nzs = -zf * sf;

    const unsigned* qcol = qweight + n;
    v8f acc = {};

    #pragma unroll 2
    for (int t = 0; t < TITERS; ++t) {
        // qweight rows for this lane group: k = kbase + t*32 + g*8 + [0..7] and kbase + t*32 + 16 + g*8 + [0..7]
        int kp0 = kslice * 128 + t * 4 + g;
        unsigned dw0 = __builtin_nontemporal_load(qcol + (size_t)kp0 * OUTF);
        unsigned dw1 = __builtin_nontemporal_load(qcol + (size_t)(kp0 + 2) * OUTF);

        // A fragment (ISA 16-bit 16x32 layout), two b128 LDS reads
        const char* abase = smem + t * 512 + (g * 8 + m8) * 16;
        v8h alo = *(const v8h*)(abase);
        v8h ahi = *(const v8h*)(abase + 256);
        v16h a;
        #pragma unroll
        for (int e = 0; e < 8; ++e) { a[e] = alo[e]; a[8 + e] = ahi[e]; }

        // B fragment: element e <-> k = g*8 + e (e<8), 16 + g*8 + (e-8) (e>=8); dequant nibble -> f16
        v16h b;
        #pragma unroll
        for (int e = 0; e < 16; ++e) {
            unsigned dw = (e < 8) ? dw0 : dw1;
            float qf = (float)((dw >> (4 * (e & 7))) & 0xFu);
            b[e] = (_Float16)__builtin_fmaf(qf, sf, nzs);
        }

        acc = __builtin_amdgcn_wmma_f32_16x16x32_f16(
            /*neg_a=*/false, a, /*neg_b=*/false, b,
            /*c_mod=*/(short)0, acc, /*reuse_a=*/false, /*reuse_b=*/false);
    }

    // D layout: lanes 0-15 hold rows M=0..7 in acc[0..7], N = lane. Rows 8-15
    // (lanes 16-31) are padding garbage -> only lanes 0-15 commit results.
    if (g == 0) {
        #pragma unroll
        for (int r = 0; r < 8; ++r)
            atomicAdd(out + r * OUTF + n, acc[r]);
    }
}

extern "C" void kernel_launch(void* const* d_in, const int* in_sizes, int n_in,
                              void* d_out, int out_size, void* d_ws, size_t ws_size,
                              hipStream_t stream) {
    const float*    x      = (const float*)d_in[0];
    const unsigned* qw     = (const unsigned*)d_in[1];
    const float*    scales = (const float*)d_in[2];
    const int*      zeros  = (const int*)d_in[3];
    const float*    bias   = (const float*)d_in[4];
    float*          out    = (float*)d_out;

    bias_init_kernel<<<(NROWS * OUTF + 255) / 256, 256, 0, stream>>>(bias, out);

    if (ws_size >= XWS_BYTES) {
        _Float16* xf16 = (_Float16*)d_ws;
        xconvert_kernel<<<16, 256, 0, stream>>>(x, xf16);
        q4gemm_wmma_kernel<true><<<NTILES * KSPLIT, 256, 16384, stream>>>(
            x, xf16, qw, scales, zeros, out);
    } else {
        q4gemm_wmma_kernel<false><<<NTILES * KSPLIT, 256, 16384, stream>>>(
            x, (const _Float16*)nullptr, qw, scales, zeros, out);
    }
}
